// MultiHeadAttention_68564857913728
// MI455X (gfx1250) — matrix operand order repaired and checked by reading, not compiled, not run on hardware
//
#include <hip/hip_runtime.h>
#include <hip/hip_bf16.h>
#include <stdint.h>

// Problem constants (match reference)
#define B_   4
#define S_   2048
#define H_   1024
#define NH_  16
#define HS_  64
// 1/sqrt(64) * log2(e): softmax done in exp2 domain (invariant after normalization)
#define QSCALE 0.18033688011112042f

typedef __bf16 bf16;
typedef __attribute__((ext_vector_type(16))) __bf16 v16bf;
typedef __attribute__((ext_vector_type(8)))  __bf16 v8bf;
typedef __attribute__((ext_vector_type(4)))  __bf16 v4bf;
typedef __attribute__((ext_vector_type(8)))  float   v8f;
typedef __attribute__((ext_vector_type(4)))  float   v4f;

static __device__ inline float fast_exp2(float x) {
#if __has_builtin(__builtin_amdgcn_exp2f)
  return __builtin_amdgcn_exp2f(x);
#else
  return exp2f(x);
#endif
}

// ---------------------------------------------------------------------------
// WMMA helpers (CDNA5 ISA 7.12.2, wave32)
// ---------------------------------------------------------------------------
static __device__ inline v8f wmma_bf16(v16bf a, v16bf b, v8f c) {
  return __builtin_amdgcn_wmma_f32_16x16x32_bf16(false, a, false, b, (short)0, c,
                                                 false, false);
}

// A (16x32): lane l = row (l&15); two contiguous 16B chunks at k = (l>>4)*8 and 16+(l>>4)*8
static __device__ inline v16bf wmma_a_rowmajor(const bf16* tile, int ld, int lane) {
  const int m = lane & 15, g = lane >> 4;
  v8bf lo = *(const v8bf*)(tile + (size_t)m * ld + g * 8);
  v8bf hi = *(const v8bf*)(tile + (size_t)m * ld + 16 + g * 8);
  return __builtin_shufflevector(lo, hi, 0,1,2,3,4,5,6,7,8,9,10,11,12,13,14,15);
}

// A fragment from an LDS tile with 40-element (80B) row stride
static __device__ inline v16bf lds_a_frag(const bf16* rowbase, int lane) {
  const int m = lane & 15, g = lane >> 4;
  const bf16* p = rowbase + m * 40;
  v8bf lo = *(const v8bf*)(p + g * 8);
  v8bf hi = *(const v8bf*)(p + 16 + g * 8);
  return __builtin_shufflevector(lo, hi, 0,1,2,3,4,5,6,7,8,9,10,11,12,13,14,15);
}

// B (32x16) from "n-major" storage Bsrc[n][k] (global or LDS), leading dim ld
static __device__ inline v16bf b_frag_nmajor(const bf16* colbase, int ld, int lane) {
  const int n = lane & 15, g = lane >> 4;
  const bf16* p = colbase + (size_t)n * ld + g * 8;
  v8bf lo = *(const v8bf*)p;
  v8bf hi = *(const v8bf*)(p + 16);
  return __builtin_shufflevector(lo, hi, 0,1,2,3,4,5,6,7,8,9,10,11,12,13,14,15);
}

// ---------------------------------------------------------------------------
// CDNA5 async copy (global -> LDS), tracked with ASYNCcnt
// ---------------------------------------------------------------------------
static __device__ inline unsigned lds_off_of(const void* p) {
  return (unsigned)(uintptr_t)p;   // LDS aperture: low 32 bits == LDS byte offset
}
static __device__ inline void async_copy_b128(unsigned lds_off, uint64_t gaddr) {
  asm volatile("global_load_async_to_lds_b128 %0, %1, off"
               :: "v"(lds_off), "v"(gaddr) : "memory");
}
static __device__ inline void wait_async_le4() {
  asm volatile("s_wait_asynccnt 0x4" ::: "memory");
}
static __device__ inline void wait_async_le0() {
  asm volatile("s_wait_asynccnt 0x0" ::: "memory");
}

// ---------------------------------------------------------------------------
// fp32 -> bf16 (4-wide)
// ---------------------------------------------------------------------------
__global__ void f32_to_bf16(const float* __restrict__ src, bf16* __restrict__ dst,
                            int n4) {
  int i = blockIdx.x * blockDim.x + threadIdx.x;
  int stride = gridDim.x * blockDim.x;
  const v4f* s = (const v4f*)src;
  v4bf* d = (v4bf*)dst;
  for (; i < n4; i += stride) d[i] = __builtin_convertvector(s[i], v4bf);
}

// ---------------------------------------------------------------------------
// GEMM: out[M,N] = A[M,K](bf16) @ W[N,K]^T(bf16) + bias(f32)
// 8 waves, block tile 128x128, wave tile 32x64, async double-buffered LDS.
// MODE 0: f32 out [M,N]           (final projection)
// MODE 1: bf16 out [B,NH,S,HS]    (K)
// MODE 2: bf16 out [B,NH,HS,S]    (V, transposed heads)
// MODE 3: MODE 1 * QSCALE         (Q, pre-scaled for exp2-domain softmax)
// ---------------------------------------------------------------------------
template <int MODE>
__global__ __launch_bounds__(256)
void gemm_bf16_wmma(const bf16* __restrict__ A, const bf16* __restrict__ W,
                    const float* __restrict__ bias, void* __restrict__ out,
                    int M, int N, int K) {
  __shared__ __align__(16) bf16 Abuf[2][128 * 40];
  __shared__ __align__(16) bf16 Bbuf[2][128 * 40];

  const int t = threadIdx.x;
  const int lane = t & 31;
  const int w = t >> 5, wm = w >> 1, wn = w & 1;
  const int row0 = blockIdx.y * 128, col0 = blockIdx.x * 128;
  const bf16* Ablk = A + (size_t)row0 * K;
  const bf16* Wblk = W + (size_t)col0 * K;

  const int srow  = t >> 1;  // staging row 0..127
  const int shalf = t & 1;   // which 32B half of the 64B row

  auto stage = [&](int bi, int kk) {
    uint64_t ga = (uint64_t)(uintptr_t)((const char*)(Ablk + (size_t)srow * K + kk) + shalf * 32);
    uint64_t gb = (uint64_t)(uintptr_t)((const char*)(Wblk + (size_t)srow * K + kk) + shalf * 32);
    unsigned la = lds_off_of(&Abuf[bi][srow * 40]) + shalf * 32;
    unsigned lb = lds_off_of(&Bbuf[bi][srow * 40]) + shalf * 32;
    async_copy_b128(la,      ga);
    async_copy_b128(la + 16, ga + 16);
    async_copy_b128(lb,      gb);
    async_copy_b128(lb + 16, gb + 16);
  };

  v8f acc[2][4];
#pragma unroll
  for (int i = 0; i < 2; ++i)
#pragma unroll
    for (int j = 0; j < 4; ++j) acc[i][j] = {};

  stage(0, 0);
  int cur = 0;
  for (int k0 = 0; k0 < K; k0 += 32) {
    const bool more = (k0 + 32) < K;
    if (more) stage(cur ^ 1, k0 + 32);       // prefetch next tile (DMA overlaps WMMA)
    if (more) wait_async_le4(); else wait_async_le0();
    __syncthreads();

    v16bf a0 = lds_a_frag(&Abuf[cur][(wm * 32     ) * 40], lane);
    v16bf a1 = lds_a_frag(&Abuf[cur][(wm * 32 + 16) * 40], lane);
#pragma unroll
    for (int j = 0; j < 4; ++j) {
      v16bf b = b_frag_nmajor(&Bbuf[cur][(wn * 64 + j * 16) * 40], 40, lane);
      acc[0][j] = wmma_bf16(a0, b, acc[0][j]);
      acc[1][j] = wmma_bf16(a1, b, acc[1][j]);
    }
    __syncthreads();
    cur ^= 1;
  }

  const int hi = lane >> 4, nn = lane & 15;
#pragma unroll
  for (int i = 0; i < 2; ++i)
#pragma unroll
    for (int j = 0; j < 4; ++j)
#pragma unroll
      for (int r = 0; r < 8; ++r) {
        int row = row0 + wm * 32 + i * 16 + r + hi * 8;
        int col = col0 + wn * 64 + j * 16 + nn;
        float v = acc[i][j][r] + bias[col];
        if (MODE == 0) {
          ((float*)out)[(size_t)row * N + col] = v;
        } else {
          if (MODE == 3) v *= QSCALE;
          int b = row / S_, s = row % S_;
          int h = col / HS_, d = col % HS_;
          bf16 bv = (bf16)v;
          if (MODE == 2)
            ((bf16*)out)[(((size_t)(b * NH_ + h)) * HS_ + d) * S_ + s] = bv;
          else
            ((bf16*)out)[(((size_t)(b * NH_ + h)) * S_ + s) * HS_ + d] = bv;
        }
      }
}

// ---------------------------------------------------------------------------
// Block-cooperative causal flash attention.
// 128-thread block = 4 waves = 128 queries of one head. Each 32-key tile
// (K 32x64, V 64x32) is async-DMA'd into LDS ONCE per block and consumed by
// all waves (4x less L2 traffic than per-wave streaming). Per wave: 32
// queries, 18 WMMAs per tile, wave-global running max in exp2 domain,
// row sums via ones-matrix WMMA.
// ---------------------------------------------------------------------------
template <bool MASKED>
static __device__ inline void attn_tile(const bf16* __restrict__ Kb,   // LDS, 32x(72)
                                        const bf16* __restrict__ Vb,   // LDS, 64x(40)
                                        bf16* __restrict__ pb,         // LDS, 32x(40), per-wave
                                        int lane, const v16bf (&aq)[2][2], v16bf onesb,
                                        v8f (&acc)[2][4], v8f (&lacc)[2], float& mrun) {
  const int hi = lane >> 4, nn = lane & 15;

  // K fragments: n = key-in-tile, k = d
  v16bf kb[2][2];
#pragma unroll
  for (int nf = 0; nf < 2; ++nf)
#pragma unroll
    for (int kk = 0; kk < 2; ++kk)
      kb[nf][kk] = b_frag_nmajor(Kb + (nf * 16) * 72 + kk * 32, 72, lane);

  // scores: s[qsub][keyfrag]. In the masked straddle tile, (qsub0,frag1) is
  // entirely above the diagonal -> skip its WMMAs, P := 0.
  v8f s[2][2];
#pragma unroll
  for (int qi = 0; qi < 2; ++qi)
#pragma unroll
    for (int nf = 0; nf < 2; ++nf) {
      if (MASKED && qi == 0 && nf == 1) { s[qi][nf] = {}; continue; }
      v8f a = {};
      a = wmma_bf16(aq[qi][0], kb[nf][0], a);
      a = wmma_bf16(aq[qi][1], kb[nf][1], a);
      s[qi][nf] = a;
    }

  if (MASKED) {
#pragma unroll
    for (int r = 0; r < 8; ++r) {
      int rl = r + hi * 8;                  // row within 16-row subtile
      s[0][0][r] = (nn <= rl) ? s[0][0][r] : -1e30f;   // rows 0..15 vs keys 0..15
      s[1][1][r] = (nn <= rl) ? s[1][1][r] : -1e30f;   // rows 16..31 vs keys 16..31
      // s[1][0]: keys 0..15 <= rows 16..31 -> never masked
    }
  }

  // ---- wave-global tile max: in-lane tree + single cross-lane reduction
  float tmax;
  {
    v8f em;
#pragma unroll
    for (int r = 0; r < 8; ++r) {
      float a = fmaxf(s[0][0][r], s[1][0][r]);
      float b = MASKED ? s[1][1][r] : fmaxf(s[0][1][r], s[1][1][r]);
      em[r] = fmaxf(a, b);
    }
    float m0 = fmaxf(fmaxf(em[0], em[1]), fmaxf(em[2], em[3]));
    float m1 = fmaxf(fmaxf(em[4], em[5]), fmaxf(em[6], em[7]));
    tmax = fmaxf(m0, m1);
    tmax = fmaxf(tmax, __shfl_xor(tmax, 1,  32));
    tmax = fmaxf(tmax, __shfl_xor(tmax, 2,  32));
    tmax = fmaxf(tmax, __shfl_xor(tmax, 4,  32));
    tmax = fmaxf(tmax, __shfl_xor(tmax, 8,  32));
    tmax = fmaxf(tmax, __shfl_xor(tmax, 16, 32));
  }

  // wave-uniform: rescale accumulators only when the running max grows
  if (tmax > mrun) {
    float fac = fast_exp2(mrun - tmax);
    mrun = tmax;
#pragma unroll
    for (int qi = 0; qi < 2; ++qi) {
      lacc[qi] *= fac;
#pragma unroll
      for (int j = 0; j < 4; ++j) acc[qi][j] *= fac;
    }
  }

  // ---- probabilities -> per-wave LDS P tile (32 rows x 32 keys, stride 40)
#pragma unroll
  for (int qi = 0; qi < 2; ++qi)
#pragma unroll
    for (int r = 0; r < 8; ++r) {
      int row = qi * 16 + r + hi * 8;
      float p0 = fast_exp2(s[qi][0][r] - mrun);
      pb[row * 40 + nn] = (bf16)p0;
      if (MASKED && qi == 0) {
        pb[row * 40 + 16 + nn] = (bf16)0.0f;
      } else {
        float p1 = fast_exp2(s[qi][1][r] - mrun);
        pb[row * 40 + 16 + nn] = (bf16)p1;
      }
    }

  // per-wave LDS region: DS in-order; wait stores, then read P in A layout
  asm volatile("s_wait_dscnt 0x0" ::: "memory");

  v16bf vb[4];
#pragma unroll
  for (int j = 0; j < 4; ++j)
    vb[j] = b_frag_nmajor(Vb + (j * 16) * 40, 40, lane);

#pragma unroll
  for (int qi = 0; qi < 2; ++qi) {
    v16bf pa = lds_a_frag(pb + (qi * 16) * 40, lane);
    lacc[qi] = wmma_bf16(pa, onesb, lacc[qi]);          // row sums
#pragma unroll
    for (int j = 0; j < 4; ++j)
      acc[qi][j] = wmma_bf16(pa, vb[j], acc[qi][j]);
  }
  asm volatile("" ::: "memory");
}

__global__ __launch_bounds__(128)
void attn_kernel(const bf16* __restrict__ Qh, const bf16* __restrict__ Kh,
                 const bf16* __restrict__ Vt, bf16* __restrict__ ctx) {
  __shared__ __align__(16) bf16 Kbuf[2][32 * 72];
  __shared__ __align__(16) bf16 Vbuf[2][64 * 40];
  __shared__ __align__(16) bf16 Pbuf[4][32 * 40];

  const int t = threadIdx.x;
  const int lane = t & 31;
  const int wv = t >> 5;                     // wave 0..3
  const int hi = lane >> 4, nn = lane & 15;
  const int Q0  = blockIdx.x * 128;          // block's query range [Q0, Q0+128)
  const int qw0 = Q0 + wv * 32;              // this wave's 32 queries
  const int bh  = blockIdx.y;                // b*NH + h

  const bf16* Qp = Qh + (size_t)bh * S_ * HS_;
  const bf16* Kp = Kh + (size_t)bh * S_ * HS_;
  const bf16* Vp = Vt + (size_t)bh * HS_ * S_;
  bf16* pb = Pbuf[wv];

  v16bf aq[2][2];
#pragma unroll
  for (int qi = 0; qi < 2; ++qi)
#pragma unroll
    for (int kk = 0; kk < 2; ++kk)
      aq[qi][kk] = wmma_a_rowmajor(Qp + (size_t)(qw0 + qi * 16) * HS_ + kk * 32, HS_, lane);

  v16bf onesb;
#pragma unroll
  for (int i = 0; i < 16; ++i) onesb[i] = (bf16)1.0f;

  v8f acc[2][4];
#pragma unroll
  for (int qi = 0; qi < 2; ++qi)
#pragma unroll
    for (int j = 0; j < 4; ++j) acc[qi][j] = {};
  v8f lacc[2] = {{}, {}};
  float mrun = -1e30f;

  // cooperative K/V staging: per thread 4 async b128 (K: 2, V: 2)
  const int krow = t >> 2, kseg = t & 3;     // K tile: 32 rows x 128B
  const int vrow = t >> 1, vseg = t & 1;     // V tile: 64 rows x 64B
  auto stage = [&](int bi, int kt) {
    uint64_t gk = (uint64_t)(uintptr_t)(Kp + (size_t)(kt + krow) * HS_ + kseg * 16);
    unsigned lk = lds_off_of(&Kbuf[bi][krow * 72 + kseg * 16]);
    async_copy_b128(lk,      gk);
    async_copy_b128(lk + 16, gk + 16);
    uint64_t gv = (uint64_t)(uintptr_t)(Vp + (size_t)vrow * S_ + kt + vseg * 16);
    unsigned lv = lds_off_of(&Vbuf[bi][vrow * 40 + vseg * 16]);
    async_copy_b128(lv,      gv);
    async_copy_b128(lv + 16, gv + 16);
  };

  const int ktend = Q0 + 128;                // block-uniform trip count
  stage(0, 0);
  int cur = 0;
  for (int kt = 0; kt < ktend; kt += 32) {
    const bool more = (kt + 32) < ktend;
    if (more) stage(cur ^ 1, kt + 32);
    if (more) wait_async_le4(); else wait_async_le0();
    __syncthreads();

    if (kt < qw0)
      attn_tile<false>(Kbuf[cur], Vbuf[cur], pb, lane, aq, onesb, acc, lacc, mrun);
    else if (kt == qw0)
      attn_tile<true >(Kbuf[cur], Vbuf[cur], pb, lane, aq, onesb, acc, lacc, mrun);

    __syncthreads();
    cur ^= 1;
  }

  const int bb = bh >> 4, h = bh & 15;
#pragma unroll
  for (int qi = 0; qi < 2; ++qi)
#pragma unroll
    for (int r = 0; r < 8; ++r) {
      float inv = 1.0f / lacc[qi][r];
      int s = qw0 + qi * 16 + r + hi * 8;
      size_t rowoff = ((size_t)bb * S_ + s) * H_;
#pragma unroll
      for (int j = 0; j < 4; ++j)
        ctx[rowoff + h * HS_ + j * 16 + nn] = (bf16)(acc[qi][j][r] * inv);
    }
}

// ---------------------------------------------------------------------------
// Launcher
// ---------------------------------------------------------------------------
extern "C" void kernel_launch(void* const* d_in, const int* in_sizes, int n_in,
                              void* d_out, int out_size, void* d_ws, size_t ws_size,
                              hipStream_t stream) {
  (void)in_sizes; (void)n_in; (void)out_size; (void)ws_size;

  const float* q  = (const float*)d_in[0];
  const float* k  = (const float*)d_in[1];
  const float* v  = (const float*)d_in[2];
  // d_in[3] = causal mask (computed analytically)
  const float* Wq = (const float*)d_in[4];
  const float* bq = (const float*)d_in[5];
  const float* Wk = (const float*)d_in[6];
  const float* bk = (const float*)d_in[7];
  const float* Wv = (const float*)d_in[8];
  const float* bv = (const float*)d_in[9];
  const float* Wo = (const float*)d_in[10];
  const float* bo = (const float*)d_in[11];

  const size_t NTOK = (size_t)B_ * S_;   // 8192
  const size_t XE   = NTOK * H_;         // 8,388,608
  const size_t WE   = (size_t)H_ * H_;

  bf16* Xq  = (bf16*)d_ws;
  bf16* Xk  = Xq  + XE;
  bf16* Xv  = Xk  + XE;
  bf16* Wqb = Xv  + XE;
  bf16* Wkb = Wqb + WE;
  bf16* Wvb = Wkb + WE;
  bf16* Wob = Wvb + WE;
  bf16* Qh  = Wob + WE;
  bf16* Kh  = Qh  + XE;
  bf16* Vt  = Kh  + XE;
  bf16* ctx = Vt  + XE;

  f32_to_bf16<<<2048, 256, 0, stream>>>(q,  Xq,  (int)(XE / 4));
  f32_to_bf16<<<2048, 256, 0, stream>>>(k,  Xk,  (int)(XE / 4));
  f32_to_bf16<<<2048, 256, 0, stream>>>(v,  Xv,  (int)(XE / 4));
  f32_to_bf16<<<256,  256, 0, stream>>>(Wq, Wqb, (int)(WE / 4));
  f32_to_bf16<<<256,  256, 0, stream>>>(Wk, Wkb, (int)(WE / 4));
  f32_to_bf16<<<256,  256, 0, stream>>>(Wv, Wvb, (int)(WE / 4));
  f32_to_bf16<<<256,  256, 0, stream>>>(Wo, Wob, (int)(WE / 4));

  dim3 gg(H_ / 128, (unsigned)(NTOK / 128));   // (8, 64)
  gemm_bf16_wmma<3><<<gg, 256, 0, stream>>>(Xq, Wqb, bq, Qh, (int)NTOK, H_, H_);
  gemm_bf16_wmma<1><<<gg, 256, 0, stream>>>(Xk, Wkb, bk, Kh, (int)NTOK, H_, H_);
  gemm_bf16_wmma<2><<<gg, 256, 0, stream>>>(Xv, Wvb, bv, Vt, (int)NTOK, H_, H_);

  dim3 ag(S_ / 128, B_ * NH_);                 // (16, 64), 128 threads
  attn_kernel<<<ag, 128, 0, stream>>>(Qh, Kh, Vt, ctx);

  gemm_bf16_wmma<0><<<gg, 256, 0, stream>>>(ctx, Wob, bo, (float*)d_out, (int)NTOK, H_, H_);
}
